// TeethHungarianMatcher_33672543601174
// MI455X (gfx1250) — compile-verified, matrix operand order checked by reading, not run on hardware
//
#include <hip/hip_runtime.h>
#include <hip/hip_bf16.h>

typedef __attribute__((ext_vector_type(16))) _Float16 v16h;
typedef __attribute__((ext_vector_type(8)))  float    v8f;

#define B_   4
#define Q_   100
#define C_   17
#define T_   16
#define HW_  65536
#define QT_  7            // ceil(Q/16) q-tiles
#define KC_  64           // split-K chunks
#define KCHUNK_ (HW_ / KC_)     // 1024
#define NSTEP_  (KCHUNK_ / 32)  // 32 WMMAs of K=32 per chunk

// workspace layout (floats)
#define WS_CC_   0                               // cost_class   [B][Q][T]          = 6400
#define WS_ATP_  (WS_CC_  + B_*Q_*T_)            // area_tgt prt [B][KC][32 lanes]  = 8192
#define WS_APP_  (WS_ATP_ + B_*KC_*32)           // area_pred prt[B][QT][KC][16]    = 28672
#define WS_ABP_  (WS_APP_ + B_*QT_*KC_*16)       // area_both prt[B][QT][KC][256]   = 458752

__device__ __forceinline__ float sigmoidf_fast(float x) {
    // v_exp_f32 + v_rcp_f32 (avoid the full IEEE divide sequence)
    return __builtin_amdgcn_rcpf(1.0f + __expf(-x));
}

// load 16 f32 in the ISA 16x32 f16-A/B striping, sigmoid+convert, accumulate sum
__device__ __forceinline__ v16h load_sigmoid_cvt(const float* __restrict__ row,
                                                 int k0, float amask, float& acc) {
    float4 t0 = *(const float4*)(row + k0);
    float4 t1 = *(const float4*)(row + k0 + 4);
    float4 t2 = *(const float4*)(row + k0 + 16);
    float4 t3 = *(const float4*)(row + k0 + 20);
    float v[16] = {t0.x, t0.y, t0.z, t0.w, t1.x, t1.y, t1.z, t1.w,
                   t2.x, t2.y, t2.z, t2.w, t3.x, t3.y, t3.z, t3.w};
    v16h h;
    float a0 = 0.0f, a1 = 0.0f;
    #pragma unroll
    for (int i = 0; i < 16; i += 2) {
        float s0 = amask * sigmoidf_fast(v[i]);
        float s1 = amask * sigmoidf_fast(v[i + 1]);
        h[i]     = (_Float16)s0;
        h[i + 1] = (_Float16)s1;
        a0 += s0;
        a1 += s1;
    }
    acc += a0 + a1;
    return h;
}

// load 16 f32 (binary mask values), convert to f16, accumulate sum
__device__ __forceinline__ v16h load_cvt(const float* __restrict__ row,
                                         int k0, float& acc) {
    float4 t0 = *(const float4*)(row + k0);
    float4 t1 = *(const float4*)(row + k0 + 4);
    float4 t2 = *(const float4*)(row + k0 + 16);
    float4 t3 = *(const float4*)(row + k0 + 20);
    float v[16] = {t0.x, t0.y, t0.z, t0.w, t1.x, t1.y, t1.z, t1.w,
                   t2.x, t2.y, t2.z, t2.w, t3.x, t3.y, t3.z, t3.w};
    v16h h;
    float a0 = 0.0f, a1 = 0.0f;
    #pragma unroll
    for (int i = 0; i < 16; i += 2) {
        h[i]     = (_Float16)v[i];
        h[i + 1] = (_Float16)v[i + 1];
        a0 += v[i];
        a1 += v[i + 1];
    }
    acc += a0 + a1;
    return h;
}

// ---------------------------------------------------------------------------
// Kernel 1: classification cost = -softmax(pred_logits)[..., labels]
// one thread per (b, q)
// ---------------------------------------------------------------------------
__global__ void teeth_cost_class_kernel(const float* __restrict__ pred_logits,
                                        const int*   __restrict__ labels,
                                        float*       __restrict__ cc) {
    int idx = blockIdx.x * blockDim.x + threadIdx.x;
    if (idx >= B_ * Q_) return;
    int b = idx / Q_;
    const float* lg = pred_logits + (size_t)idx * C_;

    float mx = lg[0];
    #pragma unroll
    for (int i = 1; i < C_; ++i) mx = fmaxf(mx, lg[i]);
    float e[C_];
    float sum = 0.0f;
    #pragma unroll
    for (int i = 0; i < C_; ++i) { e[i] = __expf(lg[i] - mx); sum += e[i]; }
    float inv = 1.0f / sum;
    #pragma unroll
    for (int t = 0; t < T_; ++t) {
        int lbl = labels[b * T_ + t];   // 0..15
        cc[(size_t)idx * T_ + t] = -e[lbl] * inv;
    }
}

// ---------------------------------------------------------------------------
// Kernel 2: split-K pairwise overlap GEMM via V_WMMA_F32_16X16X32_F16.
// grid = B * QT * KC workgroups, 32 threads (1 wave) each.
// A = sigmoid(pred_masks) rows (16 q's), B = masks rows (16 t's), K = 1024.
// Two independent accumulators break the WMMA->WMMA RAW chain (no v_nops).
// Also produces per-chunk row sums for area_pred / area_tgt.
// ---------------------------------------------------------------------------
__global__ __launch_bounds__(32)
void teeth_overlap_wmma_kernel(const float* __restrict__ pred_masks,
                               const float* __restrict__ masks,
                               float* __restrict__ abp,   // [B][QT][KC][256]
                               float* __restrict__ app,   // [B][QT][KC][16]
                               float* __restrict__ atp) { // [B][KC][32]
    const int lane = threadIdx.x;        // 0..31
    const int blk  = blockIdx.x;
    const int kc   = blk % KC_;
    const int qt   = (blk / KC_) % QT_;
    const int b    = blk / (KC_ * QT_);

    const int hi = lane >> 4;            // 0: K-groups {0..7,16..23}; 1: {8..15,24..31}
    const int r  = lane & 15;            // tile row (q) / tile col (t)
    const int q  = qt * 16 + r;
    const bool qok = (q < Q_);
    const float amask = qok ? 1.0f : 0.0f;

    const float* Arow = pred_masks + (size_t)(b * Q_ + (qok ? q : 0)) * HW_;
    const float* Brow = masks      + (size_t)(b * T_ + r) * HW_;

    v8f c0 = {0.f, 0.f, 0.f, 0.f, 0.f, 0.f, 0.f, 0.f};
    v8f c1 = {0.f, 0.f, 0.f, 0.f, 0.f, 0.f, 0.f, 0.f};
    float asum = 0.0f;
    float bsum = 0.0f;

    const int kbase = kc * KCHUNK_ + hi * 8;

    for (int s = 0; s < NSTEP_; s += 2) {
        const int k0 = kbase + s * 32;
        const int k1 = k0 + 32;

        // prefetch ahead (global_prefetch_b8)
        __builtin_prefetch(Arow + k0 + 512, 0, 1);
        __builtin_prefetch(Brow + k0 + 512, 0, 1);

        v16h a0 = load_sigmoid_cvt(Arow, k0, amask, asum);
        v16h b0 = load_cvt(Brow, k0, bsum);
        c0 = __builtin_amdgcn_wmma_f32_16x16x32_f16(
                false, a0, false, b0, (short)0, c0, false, false);

        v16h a1 = load_sigmoid_cvt(Arow, k1, amask, asum);
        v16h b1 = load_cvt(Brow, k1, bsum);
        c1 = __builtin_amdgcn_wmma_f32_16x16x32_f16(
                false, a1, false, b1, (short)0, c1, false, false);
    }

    // area_pred partial: row sums (lane r + lane r+16 hold the two K halves)
    asum += __shfl_xor(asum, 16, 32);
    if (hi == 0)
        app[(size_t)(((b * QT_ + qt) * KC_) + kc) * 16 + r] = asum;

    // area_tgt partial: only one q-tile's workgroups need to emit it
    if (qt == 0)
        atp[(size_t)(b * KC_ + kc) * 32 + lane] = bsum;

    // write the 16x16 f32 accumulator tile
    // lanes 0-15: VGPR i -> (M=i,   N=lane); lanes 16-31: VGPR i -> (M=i+8, N=lane-16)
    float* cb = abp + (size_t)(((b * QT_ + qt) * KC_) + kc) * 256;
    #pragma unroll
    for (int i = 0; i < 8; ++i) {
        int m = i + hi * 8;
        cb[m * 16 + r] = c0[i] + c1[i];
    }
}

// ---------------------------------------------------------------------------
// Kernel 3: reduce split-K partials, assemble IoU/dice/class cost matrix.
// one thread per (b, q, t)
// ---------------------------------------------------------------------------
__global__ void teeth_combine_kernel(const float* __restrict__ cc,
                                     const float* __restrict__ abp,
                                     const float* __restrict__ app,
                                     const float* __restrict__ atp,
                                     const int*   __restrict__ valid_masks,
                                     float*       __restrict__ out) {
    int idx = blockIdx.x * blockDim.x + threadIdx.x;
    if (idx >= B_ * Q_ * T_) return;
    int t = idx % T_;
    int q = (idx / T_) % Q_;
    int b = idx / (T_ * Q_);
    int qt = q >> 4;
    int m  = q & 15;

    const float* ab_base = abp + (size_t)((b * QT_ + qt) * KC_) * 256 + m * 16 + t;
    const float* ap_base = app + (size_t)((b * QT_ + qt) * KC_) * 16 + m;
    const float* at_base = atp + (size_t)(b * KC_) * 32;

    float ab = 0.0f, ap = 0.0f, at = 0.0f;
    for (int kc = 0; kc < KC_; ++kc) {
        ab += ab_base[kc * 256];
        ap += ap_base[kc * 16];
        at += at_base[kc * 32 + t] + at_base[kc * 32 + t + 16];
    }

    float uni  = ap + at - ab;
    float cost_mask = 1.0f - ab / (uni + 1e-6f);
    float cost_dice = 1.0f - 2.0f * ab / (ap + at + 1e-6f);
    float v = cc[idx] + 2.0f * cost_mask + 1.0f * cost_dice;  // COST_CLASS=1, COST_MASK=2, COST_DICE=1
    out[idx] = (valid_masks[b * T_ + t] != 0) ? v : 1e6f;
}

// ---------------------------------------------------------------------------
extern "C" void kernel_launch(void* const* d_in, const int* in_sizes, int n_in,
                              void* d_out, int out_size, void* d_ws, size_t ws_size,
                              hipStream_t stream) {
    const float* pred_logits = (const float*)d_in[0];   // [B,Q,17]
    const float* pred_masks  = (const float*)d_in[1];   // [B,Q,256,256]
    const int*   labels      = (const int*)  d_in[2];   // [B,T]
    const float* masks       = (const float*)d_in[3];   // [B,T,256,256]
    const int*   valid_masks = (const int*)  d_in[4];   // [B,T]
    float* out = (float*)d_out;                          // [B,Q,T]

    float* ws  = (float*)d_ws;
    float* cc  = ws + WS_CC_;
    float* atp = ws + WS_ATP_;
    float* app = ws + WS_APP_;
    float* abp = ws + WS_ABP_;

    // 1) classification cost (tiny)
    teeth_cost_class_kernel<<<(B_ * Q_ + 255) / 256, 256, 0, stream>>>(pred_logits, labels, cc);

    // 2) pairwise overlap GEMM, split-K, one wave per 16x16 tile-chunk
    teeth_overlap_wmma_kernel<<<B_ * QT_ * KC_, 32, 0, stream>>>(pred_masks, masks, abp, app, atp);

    // 3) reduce partials + final cost assembly
    teeth_combine_kernel<<<(B_ * Q_ * T_ + 255) / 256, 256, 0, stream>>>(cc, abp, app, atp,
                                                                         valid_masks, out);
}
